// HAN_23768349016456
// MI455X (gfx1250) — compile-verified, matrix operand order checked by reading, not run on hardware
//
#include <hip/hip_runtime.h>
#include <hip/hip_bf16.h>
#include <math.h>

// ---------------- problem constants (match reference) ----------------
#define N_FILE 50000
#define N_AUTH 20000
#define NEDGE  400000
#define HEADS  4
#define DHID   64
#define HID    256   // HEADS*DHID
#define DOUT3  64
#define NEG_SLOPE 0.2f

// ---------------- WMMA types ----------------
typedef __attribute__((ext_vector_type(16))) __bf16 v16bf;
typedef __attribute__((ext_vector_type(2)))  __bf16 v2bf;
typedef __attribute__((ext_vector_type(8)))  float  v8f;

union BF16Frag { v16bf v; unsigned u32[8]; };

__device__ __forceinline__ unsigned f2bf_rne(float x) {
    unsigned u = __float_as_uint(x);
    u += 0x7FFFu + ((u >> 16) & 1u);          // round-to-nearest-even
    return u >> 16;
}
__device__ __forceinline__ unsigned packbf(float lo, float hi) {
#if __has_builtin(__builtin_amdgcn_cvt_pk_bf16_f32)
    v2bf p = __builtin_amdgcn_cvt_pk_bf16_f32(lo, hi);
    return __builtin_bit_cast(unsigned, p);
#else
    return (f2bf_rne(hi) << 16) | f2bf_rne(lo);
#endif
}

// ---------------- weight prepack: fp32 [K,M] -> bf16 in WMMA B-fragment order
// tile (kb, nt): 512 bf16 = 256 u32; lane L holds col n = nt*16 + (L&15),
// u32 j holds k = kb*32 + ((j&4)?16:0) + (L>>4)*8 + 2*(j&3) (lo) and k+1 (hi)
__global__ __launch_bounds__(256)
void han_pack_w(const float* __restrict__ Wm, unsigned* __restrict__ pb, int K, int M) {
    int i = blockIdx.x * 256 + threadIdx.x;
    int total = (K * M) >> 1;
    if (i >= total) return;
    int j    = i & 7;
    int lane = (i >> 3) & 31;
    int rest = i >> 8;
    int ntiles = M >> 4;
    int nt = rest % ntiles;
    int kb = rest / ntiles;
    int n = nt * 16 + (lane & 15);
    int k = kb * 32 + ((j & 4) ? 16 : 0) + (lane >> 4) * 8 + 2 * (j & 3);
    pb[i] = packbf(Wm[(size_t)k * M + n], Wm[(size_t)(k + 1) * M + n]);
}

// ---------------- bf16 WMMA GEMM: C[N, NSUB*16] = A[N,K] * B ----------------
// One block owns 128 rows x the FULL output width (A read exactly once).
// 8 waves x 16 rows; each wave carries NSUB 16x16 f32 accumulator tiles.
#define BM 128

template <int NSUB>
__global__ __launch_bounds__(256)
void han_gemm_bf16_wmma(const float* __restrict__ A, const unsigned* __restrict__ pB,
                        float* __restrict__ C, int N, int K) {
    const int M    = NSUB * 16;
    const int tid  = threadIdx.x;
    const int wave = tid >> 5;
    const int lane = tid & 31;
    const int half = lane >> 4;      // 0: lanes 0-15, 1: lanes 16-31
    const int r    = lane & 15;
    const int row0 = blockIdx.x * BM;

    const int  grow = row0 + wave * 16 + r;          // A row this lane feeds
    const bool inb  = grow < N;
    const unsigned mz = inb ? 0xFFFFFFFFu : 0u;      // branchless row mask
    const float* arow = A + (size_t)(inb ? grow : (N - 1)) * K;

    v8f acc[NSUB];
    #pragma unroll
    for (int s = 0; s < NSUB; ++s) acc[s] = v8f{};

    for (int k0 = 0; k0 < K; k0 += 32) {
        // A fragment: two contiguous 8-float runs per lane -> 4x b128 loads
        const float4* p0 = (const float4*)(arow + k0 + half * 8);
        const float4* p1 = (const float4*)(arow + k0 + 16 + half * 8);
        float4 a0 = p0[0], a1 = p0[1];
        float4 b0 = p1[0], b1 = p1[1];
        BF16Frag af;
        af.u32[0] = packbf(a0.x, a0.y) & mz;
        af.u32[1] = packbf(a0.z, a0.w) & mz;
        af.u32[2] = packbf(a1.x, a1.y) & mz;
        af.u32[3] = packbf(a1.z, a1.w) & mz;
        af.u32[4] = packbf(b0.x, b0.y) & mz;
        af.u32[5] = packbf(b0.z, b0.w) & mz;
        af.u32[6] = packbf(b1.x, b1.y) & mz;
        af.u32[7] = packbf(b1.z, b1.w) & mz;

        const unsigned* bbase =
            pB + (((size_t)(k0 >> 5) * NSUB) << 8) + lane * 8;
        #pragma unroll
        for (int s = 0; s < NSUB; ++s) {
            const uint4* bp = (const uint4*)(bbase + (s << 8));
            uint4 q0 = bp[0], q1 = bp[1];
            BF16Frag bf;
            bf.u32[0] = q0.x; bf.u32[1] = q0.y; bf.u32[2] = q0.z; bf.u32[3] = q0.w;
            bf.u32[4] = q1.x; bf.u32[5] = q1.y; bf.u32[6] = q1.z; bf.u32[7] = q1.w;
            acc[s] = __builtin_amdgcn_wmma_f32_16x16x32_bf16(
                false, af.v, false, bf.v, (short)0, acc[s], false, false);
        }
    }

    // store D: VGPR v -> row (wave*16 + half*8 + v), col = s*16 + r
    #pragma unroll
    for (int v = 0; v < 8; ++v) {
        int orow = row0 + wave * 16 + half * 8 + v;
        if (orow < N) {
            float* crow = C + (size_t)orow * M + r;
            #pragma unroll
            for (int s = 0; s < NSUB; ++s) crow[s * 16] = acc[s][v];
        }
    }
}

// ---------------- el[n,h] = sum_d fs[n, h*dout+d] * attn[h,d] --------------
__global__ __launch_bounds__(256)
void han_rowdot(const float* __restrict__ F, const float* __restrict__ attn,
                float* __restrict__ out, int npairs, int heads, int dout) {
    int pair = blockIdx.x * 8 + (threadIdx.x >> 5);
    int lane = threadIdx.x & 31;
    if (pair >= npairs) return;
    int n = pair / heads, h = pair - n * heads;
    const float* fp = F + (size_t)n * heads * dout + h * dout;
    const float* ap = attn + h * dout;
    float s = 0.0f;
    for (int d = lane; d < dout; d += 32) s += fp[d] * ap[d];
    #pragma unroll
    for (int off = 16; off; off >>= 1) s += __shfl_xor(s, off, 32);
    if (lane == 0) out[pair] = s;
}

// ---------------- fill ------------------------------------------------------
__global__ void han_fill(float* __restrict__ p, float v, long long n) {
    long long i = (long long)blockIdx.x * 256 + threadIdx.x;
    if (i < n) p[i] = v;
}

// ---------------- float atomic max (bit trick; init must be -inf) ----------
__device__ __forceinline__ void atomicMaxF(float* addr, float val) {
    if (val >= 0.0f) atomicMax((int*)addr, __float_as_int(val));
    else             atomicMin((unsigned int*)addr, (unsigned int)__float_as_int(val));
}

// ---------------- edge pass 1: e = lrelu(el[src]+er[dst]); m = segmax ------
__global__ __launch_bounds__(256)
void han_edge_max(const int* __restrict__ src, const int* __restrict__ dst,
                  const float* __restrict__ el, const float* __restrict__ er,
                  float* __restrict__ ebuf, float* __restrict__ m,
                  int nedge, int heads) {
    int i = blockIdx.x * 256 + threadIdx.x;
    if (i >= nedge * heads) return;
    int e = i / heads, h = i - e * heads;
    int s = src[e], d = dst[e];
    float v = el[s * heads + h] + er[d * heads + h];
    v = (v > 0.0f) ? v : NEG_SLOPE * v;
    ebuf[i] = v;
    atomicMaxF(&m[d * heads + h], v);
}

// ---------------- edge pass 2: alpha = exp(e - m[dst]); denom = segsum -----
__global__ __launch_bounds__(256)
void han_edge_alpha(const int* __restrict__ dst, const float* __restrict__ ebuf,
                    const float* __restrict__ m, float* __restrict__ alpha,
                    float* __restrict__ denom, int nedge, int heads) {
    int i = blockIdx.x * 256 + threadIdx.x;
    if (i >= nedge * heads) return;
    int e = i / heads, h = i - e * heads;
    int d = dst[e];
    float a = expf(ebuf[i] - m[d * heads + h]);
    alpha[i] = a;
    atomicAdd(&denom[d * heads + h], a);
}

// ---------------- edge pass 3: rst[dst] += fs[src] * alpha/denom -----------
// one block per edge; float4 gathers, per-dword f32 atomics into rst
__global__ __launch_bounds__(64)
void han_edge_agg(const int* __restrict__ src, const int* __restrict__ dst,
                  const float* __restrict__ fs, const float* __restrict__ alpha,
                  const float* __restrict__ denom, float* __restrict__ rst,
                  int heads, int dout) {
    int e = blockIdx.x;
    int s = src[e], d = dst[e];
    int M = heads * dout;
    for (int i = threadIdx.x * 4; i < M; i += 256) {
        int h = i / dout;
        float a = alpha[e * heads + h] / denom[d * heads + h];
        float4 v = *(const float4*)(fs + (size_t)s * M + i);
        float* rp = rst + (size_t)d * M + i;
        atomicAdd(rp + 0, v.x * a);
        atomicAdd(rp + 1, v.y * a);
        atomicAdd(rp + 2, v.z * a);
        atomicAdd(rp + 3, v.w * a);
    }
}

// ---------------- epilogue: out = elu(rst + bias) --------------------------
__global__ __launch_bounds__(256)
void han_elu(const float* __restrict__ rst, const float* __restrict__ bias,
             float* __restrict__ out, long long total, int m) {
    long long i = (long long)blockIdx.x * 256 + threadIdx.x;
    if (i >= total) return;
    int c = (int)(i % m);
    float x = rst[i] + bias[c];
    out[i] = (x > 0.0f) ? x : (expf(x) - 1.0f);
}

// ---------------- host-side: one full bipartite GAT conv -------------------
struct GatScratch {
    float *fs, *fd, *el, *er, *m, *den, *ebuf, *alpha, *rst;
    unsigned *pbA, *pbB;
};

static void launch_gemm(hipStream_t st, const float* A, const unsigned* pB,
                        float* C, int N, int K, int M) {
    dim3 grid((N + BM - 1) / BM);
    if (M == 256)     han_gemm_bf16_wmma<16><<<grid, 256, 0, st>>>(A, pB, C, N, K);
    else /* M == 64 */ han_gemm_bf16_wmma<4><<<grid, 256, 0, st>>>(A, pB, C, N, K);
}

static void run_gat(hipStream_t st,
                    const float* hs, int Ns, int din_s,
                    const float* hd, int Nd, int din_d,
                    const int* src, const int* dst,
                    const float* Wsrc, const float* Wdst,
                    const float* attn_l, const float* attn_r, const float* bias,
                    int heads, int dout, float* out, const GatScratch& w) {
    const int M = heads * dout;
    // prepack weights into WMMA fragment order (bf16)
    han_pack_w<<<((din_s * M / 2) + 255) / 256, 256, 0, st>>>(Wsrc, w.pbA, din_s, M);
    han_pack_w<<<((din_d * M / 2) + 255) / 256, 256, 0, st>>>(Wdst, w.pbB, din_d, M);
    // dense projections (WMMA)
    launch_gemm(st, hs, w.pbA, w.fs, Ns, din_s, M);
    launch_gemm(st, hd, w.pbB, w.fd, Nd, din_d, M);
    // attention row dots
    int pl = Ns * heads, pr = Nd * heads;
    han_rowdot<<<(pl + 7) / 8, 256, 0, st>>>(w.fs, attn_l, w.el, pl, heads, dout);
    han_rowdot<<<(pr + 7) / 8, 256, 0, st>>>(w.fd, attn_r, w.er, pr, heads, dout);
    // init segment buffers
    long long nm = (long long)Nd * heads;
    long long nr = (long long)Nd * M;
    han_fill<<<(unsigned)((nm + 255) / 256), 256, 0, st>>>(w.m,   -INFINITY, nm);
    han_fill<<<(unsigned)((nm + 255) / 256), 256, 0, st>>>(w.den, 0.0f,      nm);
    han_fill<<<(unsigned)((nr + 255) / 256), 256, 0, st>>>(w.rst, 0.0f,      nr);
    // edge softmax + aggregation
    int eh = NEDGE * heads;
    han_edge_max  <<<(eh + 255) / 256, 256, 0, st>>>(src, dst, w.el, w.er, w.ebuf, w.m, NEDGE, heads);
    han_edge_alpha<<<(eh + 255) / 256, 256, 0, st>>>(dst, w.ebuf, w.m, w.alpha, w.den, NEDGE, heads);
    han_edge_agg  <<<NEDGE, 64, 0, st>>>(src, dst, w.fs, w.alpha, w.den, w.rst, heads, dout);
    // activation
    long long total = (long long)Nd * M;
    han_elu<<<(unsigned)((total + 255) / 256), 256, 0, st>>>(w.rst, bias, out, total, M);
}

// ---------------- entry ----------------------------------------------------
extern "C" void kernel_launch(void* const* d_in, const int* in_sizes, int n_in,
                              void* d_out, int out_size, void* d_ws, size_t ws_size,
                              hipStream_t stream) {
    // inputs (setup_inputs() insertion order)
    const float* xf     = (const float*)d_in[0];   // [50000,256]
    const float* xa     = (const float*)d_in[1];   // [20000,128]
    const int*   fa_src = (const int*)d_in[2];
    const int*   fa_dst = (const int*)d_in[3];
    const int*   af_src = (const int*)d_in[4];
    const int*   af_dst = (const int*)d_in[5];
    // params: 5 convs x (W_src, W_dst, attn_l, attn_r, bias)
    const float* P[25];
    for (int i = 0; i < 25; ++i) P[i] = (const float*)d_in[6 + i];
    const float** l1fa = &P[0];
    const float** l1af = &P[5];
    const float** l2fa = &P[10];
    const float** l2af = &P[15];
    const float** l3af = &P[20];

    // workspace carving (floats; every chunk is a multiple of 4 -> 16B aligned)
    float* W = (float*)d_ws;
    size_t o = 0;
    auto carve = [&](size_t n) { float* p = W + o; o += n; return p; };
    float* f1 = carve((size_t)N_FILE * HID);
    float* a1 = carve((size_t)N_AUTH * HID);
    float* f2 = carve((size_t)N_FILE * HID);
    float* a2 = carve((size_t)N_AUTH * HID);
    GatScratch ws;
    ws.fs    = carve((size_t)N_FILE * HID);
    ws.fd    = carve((size_t)N_FILE * HID);
    ws.rst   = carve((size_t)N_FILE * HID);
    ws.el    = carve((size_t)N_FILE * HEADS);
    ws.er    = carve((size_t)N_FILE * HEADS);
    ws.m     = carve((size_t)N_FILE * HEADS);
    ws.den   = carve((size_t)N_FILE * HEADS);
    ws.ebuf  = carve((size_t)NEDGE  * HEADS);
    ws.alpha = carve((size_t)NEDGE  * HEADS);
    ws.pbA   = (unsigned*)carve((size_t)HID * HID / 2);  // max K*M/2 u32
    ws.pbB   = (unsigned*)carve((size_t)HID * HID / 2);

    // layer 1
    run_gat(stream, xf, N_FILE, 256, xa, N_AUTH, 128, fa_src, fa_dst,
            l1fa[0], l1fa[1], l1fa[2], l1fa[3], l1fa[4], HEADS, DHID, a1, ws);
    run_gat(stream, xa, N_AUTH, 128, xf, N_FILE, 256, af_src, af_dst,
            l1af[0], l1af[1], l1af[2], l1af[3], l1af[4], HEADS, DHID, f1, ws);
    // layer 2
    run_gat(stream, f1, N_FILE, HID, a1, N_AUTH, HID, fa_src, fa_dst,
            l2fa[0], l2fa[1], l2fa[2], l2fa[3], l2fa[4], HEADS, DHID, a2, ws);
    run_gat(stream, a1, N_AUTH, HID, f1, N_FILE, HID, af_src, af_dst,
            l2af[0], l2af[1], l2af[2], l2af[3], l2af[4], HEADS, DHID, f2, ws);
    // layer 3 -> d_out [50000, 64]
    run_gat(stream, a2, N_AUTH, HID, f2, N_FILE, HID, af_src, af_dst,
            l3af[0], l3af[1], l3af[2], l3af[3], l3af[4], 1, DOUT3, (float*)d_out, ws);
}
